// CausalAttention_22462678958384
// MI455X (gfx1250) — compile-verified, matrix operand order checked by reading, not run on hardware
//
#include <hip/hip_runtime.h>
#include <hip/hip_bf16.h>

typedef _Float16 v16h __attribute__((ext_vector_type(16)));
typedef _Float16 v8h  __attribute__((ext_vector_type(8)));
typedef float    v8f  __attribute__((ext_vector_type(8)));
typedef int      v4i  __attribute__((ext_vector_type(4)));

#define DIN   1024
#define HD    128
#define SEQ   4096
#define NB    4
#define MTOT  (NB*SEQ)            // 16384 rows total
#define QKV_BYTES ((size_t)MTOT*HD*2)   // 4 MB per f16 [M,H] buffer
#define WFRAG_HALVES 131072             // halves per repacked weight (256 tiles * 512)

#if __has_builtin(__builtin_amdgcn_global_load_async_to_lds_b128) && \
    __has_builtin(__builtin_amdgcn_s_wait_asynccnt)
#define USE_ASYNC_LDS 1
#else
#define USE_ASYNC_LDS 0
#endif

#if USE_ASYNC_LDS
#define GPTR(p) ((__attribute__((address_space(1))) v4i*)(p))
#define LPTR(p) ((__attribute__((address_space(3))) v4i*)(p))
#endif

// K index inside a 16x32 f16 A/B fragment for (lane, half) per CDNA5 ISA 7.12.2
__device__ __forceinline__ int kidx(int lane, int h) {
    return 16 * (h >> 3) + ((lane & 16) ? 8 : 0) + (h & 7);
}

__device__ __forceinline__ v8f wmma_f16(v16h a, v16h b, v8f c) {
    return __builtin_amdgcn_wmma_f32_16x16x32_f16(false, a, false, b, (short)0, c, false, false);
}

// ---------------------------------------------------------------------------
// Kernel 0: repack Wq/Wk/Wv (128x1024) and Wo (1024x128) fp32 -> f16 B-frag order
//   frag addr = ((nt*KT + kt)*32 + lane)*16 + h ;  element = W[n*Kdim + k]
// ---------------------------------------------------------------------------
__global__ __launch_bounds__(256) void repack_kernel(
    const float* __restrict__ Wq, const float* __restrict__ Wk,
    const float* __restrict__ Wv, const float* __restrict__ Wo,
    _Float16* __restrict__ WqF, _Float16* __restrict__ WkF,
    _Float16* __restrict__ WvF, _Float16* __restrict__ WoF)
{
    int gid   = blockIdx.x * 256 + threadIdx.x;     // 4*131072 total
    int widx  = gid >> 17;
    int local = gid & (WFRAG_HALVES - 1);
    int h     = local & 15;
    int lane  = (local >> 4) & 31;
    int tile  = local >> 9;
    int kit   = kidx(lane, h);

    float val;
    _Float16* dst;
    if (widx < 3) {                                  // Wq/Wk/Wv: Kdim=1024, KT=32
        int kt = tile & 31, nt = tile >> 5;
        int n = nt * 16 + (lane & 15);
        int k = kt * 32 + kit;
        const float* W = (widx == 0) ? Wq : (widx == 1) ? Wk : Wv;
        val = W[n * DIN + k];
        dst = (widx == 0) ? WqF : (widx == 1) ? WkF : WvF;
    } else {                                         // Wo: Kdim=128, KT=4
        int kt = tile & 3, nt = tile >> 2;
        int n = nt * 16 + (lane & 15);
        int k = kt * 32 + kit;
        val = Wo[n * HD + k];
        dst = WoF;
    }
    dst[local] = (_Float16)val;
}

// ---------------------------------------------------------------------------
// Kernel 1: fused QKV projection.  Block = 256 thr (8 waves), 16-row x tile.
// Wave w computes heads [16w,16w+16) for Q,K,V.  Q scaled by H^-1/2, V stored
// transposed [b, h, s] for the attention kernel's PV B-frag staging.
// ---------------------------------------------------------------------------
__global__ __launch_bounds__(256) void qkv_kernel(
    const float* __restrict__ x,
    const float* __restrict__ bq, const float* __restrict__ bk, const float* __restrict__ bv,
    const _Float16* __restrict__ WqF, const _Float16* __restrict__ WkF, const _Float16* __restrict__ WvF,
    _Float16* __restrict__ Qh, _Float16* __restrict__ Kh, _Float16* __restrict__ Vt)
{
    __shared__ __align__(32) _Float16 aTile[512];     // 16x32 in A-frag order [lane][h]

    const int t    = threadIdx.x;
    const int lane = t & 31;
    const int wave = t >> 5;
    const int m0   = blockIdx.x * 16;
    const int n0   = wave * 16;

    // staging map: thread t loads x[row, col..col+1] -> aTile frag slots
    const int col   = (2 * t) & 31;
    const int row   = t >> 4;
    const int slane = row + (((col >> 3) & 1) ? 16 : 0);
    const int sh    = ((col >> 4) ? 8 : 0) + (col & 7);

    v8f aq = {0,0,0,0,0,0,0,0}, ak = {0,0,0,0,0,0,0,0}, av = {0,0,0,0,0,0,0,0};

    for (int kt = 0; kt < 32; ++kt) {
        __syncthreads();
        float2 xv = *(const float2*)(x + (size_t)(m0 + row) * DIN + kt * 32 + col);
        aTile[slane * 16 + sh]     = (_Float16)xv.x;
        aTile[slane * 16 + sh + 1] = (_Float16)xv.y;
        __syncthreads();

        v16h a = *(const v16h*)(aTile + lane * 16);
        size_t boff = ((size_t)wave * 32 + kt) * 512 + lane * 16;
        v16h fq = *(const v16h*)(WqF + boff);
        v16h fk = *(const v16h*)(WkF + boff);
        v16h fv = *(const v16h*)(WvF + boff);
        aq = wmma_f16(a, fq, aq);
        ak = wmma_f16(a, fk, ak);
        av = wmma_f16(a, fv, av);
    }

    const int head = n0 + (lane & 15);
    const int mofs = (lane & 16) ? 8 : 0;
    const float SCALE = 0.08838834764831845f;        // 128^-0.5
    const float cq = bq[head], ck = bk[head], cv = bv[head];
    for (int r = 0; r < 8; ++r) {
        int m = m0 + r + mofs;
        int b = m >> 12, s = m & (SEQ - 1);
        Qh[(size_t)m * HD + head] = (_Float16)((aq[r] + cq) * SCALE);
        Kh[(size_t)m * HD + head] = (_Float16)(ak[r] + ck);
        Vt[((size_t)b * HD + head) * SEQ + s] = (_Float16)(av[r] + cv);
    }
}

// ---------------------------------------------------------------------------
// Kernel 2: causal flash attention. Block = 256 thr (8 waves), 128 query rows;
// wave w owns queries [q0, q0+16).  Key tiles of 32; K,V staged to LDS frags
// via GLOBAL_LOAD_ASYNC_TO_LDS_B128 (ASYNCcnt) when available.
// ---------------------------------------------------------------------------
__global__ __launch_bounds__(256) void attn_kernel(
    const _Float16* __restrict__ Qh, const _Float16* __restrict__ Kh,
    const _Float16* __restrict__ Vt, _Float16* __restrict__ AO)
{
    __shared__ __align__(32) _Float16 kFrag[2][4][512];  // [ntile][kstep][lane*16+h]
    __shared__ __align__(32) _Float16 vFrag[8][512];     // [head ntile][lane*16+h]
    __shared__ __align__(32) _Float16 pBuf[8][512];      // per-wave 16x32 P transpose

    const int t    = threadIdx.x;
    const int lane = t & 31;
    const int wave = t >> 5;
    const int b    = blockIdx.x >> 5;
    const int q0w  = (blockIdx.x & 31) * 128;
    const int q0   = q0w + wave * 16;

    const _Float16* Qb = Qh + (size_t)b * SEQ * HD;
    const _Float16* Kb = Kh + (size_t)b * SEQ * HD;
    const _Float16* Vb = Vt + (size_t)b * HD * SEQ;

    const int base = (lane & 16) ? 8 : 0;
    const int mofs = (lane & 16) ? 8 : 0;

    // Q fragments (A-layout) loaded straight from row-major f16
    v16h qf[4];
    {
        int m = q0 + (lane & 15);
        for (int ks = 0; ks < 4; ++ks) {
            v8h lo = *(const v8h*)(Qb + (size_t)m * HD + ks * 32 + base);
            v8h hi = *(const v8h*)(Qb + (size_t)m * HD + ks * 32 + 16 + base);
            for (int i = 0; i < 8; ++i) { qf[ks][i] = lo[i]; qf[ks][8 + i] = hi[i]; }
        }
    }

    // per-thread staging coordinates (invariant across key tiles)
    const int kgrp[2] = { (t) & 1, (t + 256) & 1 };
    int ksl[2], kks[2], knt[2], vgrp[2], vsl[2], vnt[2];
    for (int j = 0; j < 2; ++j) {
        int idx = t + j * 256;
        ksl[j] = (idx >> 1) & 31; kks[j] = (idx >> 6) & 3; knt[j] = idx >> 8;
        vgrp[j] = idx & 1; vsl[j] = (idx >> 1) & 31; vnt[j] = idx >> 6;
    }

    v8f o[8];
    float mrow[8], lrow[8];
    for (int j = 0; j < 8; ++j) for (int r = 0; r < 8; ++r) o[j][r] = 0.f;
    for (int r = 0; r < 8; ++r) { mrow[r] = -1e30f; lrow[r] = 0.f; }

    const int ktmax = (q0w + 127) >> 5;              // inclusive, uniform per block
    const float L2E = 1.44269504f;

    for (int kt = 0; kt <= ktmax; ++kt) {
        const int k0 = kt * 32;
        __syncthreads();
        // stage K B-frags: 512 x 16B copies, 2 per thread
        for (int j = 0; j < 2; ++j) {
            int key  = k0 + knt[j] * 16 + (ksl[j] & 15);
            int head = kks[j] * 32 + kgrp[j] * 16 + ((ksl[j] & 16) ? 8 : 0);
            const _Float16* src = Kb + (size_t)key * HD + head;
            _Float16* dstp = &kFrag[knt[j]][kks[j]][ksl[j] * 16 + kgrp[j] * 8];
#if USE_ASYNC_LDS
            __builtin_amdgcn_global_load_async_to_lds_b128(GPTR(src), LPTR(dstp), 0, 0);
#else
            *(v8h*)dstp = *(const v8h*)src;
#endif
            if (kt < ktmax) __builtin_prefetch(src + 32 * HD, 0, 1);
        }
        // stage V B-frags (contiguous thanks to V^T layout): 512 x 16B, 2 per thread
        for (int j = 0; j < 2; ++j) {
            int head = vnt[j] * 16 + (vsl[j] & 15);
            int key  = k0 + vgrp[j] * 16 + ((vsl[j] & 16) ? 8 : 0);
            const _Float16* src = Vb + (size_t)head * SEQ + key;
            _Float16* dstp = &vFrag[vnt[j]][vsl[j] * 16 + vgrp[j] * 8];
#if USE_ASYNC_LDS
            __builtin_amdgcn_global_load_async_to_lds_b128(GPTR(src), LPTR(dstp), 0, 0);
#else
            *(v8h*)dstp = *(const v8h*)src;
#endif
            if (kt < ktmax) __builtin_prefetch(src + 32, 0, 1);
        }
#if USE_ASYNC_LDS
        __builtin_amdgcn_s_wait_asynccnt(0);
#endif
        __syncthreads();

        if (k0 <= q0 + 15) {                          // causal: this wave needs this tile
            // scores: 2 N-tiles x 4 K-steps = 8 WMMAs
            v8f s[2];
            for (int nt = 0; nt < 2; ++nt) {
                for (int r = 0; r < 8; ++r) s[nt][r] = 0.f;
                for (int ks = 0; ks < 4; ++ks)
                    s[nt] = wmma_f16(qf[ks], *(const v16h*)(&kFrag[nt][ks][lane * 16]), s[nt]);
            }
            // causal mask (diagonal tiles only)
            if (k0 + 31 > q0) {
                for (int nt = 0; nt < 2; ++nt)
                    for (int r = 0; r < 8; ++r) {
                        int key = k0 + nt * 16 + (lane & 15);
                        int q   = q0 + r + mofs;
                        if (key > q) s[nt][r] = -1e30f;
                    }
            }
            // online softmax per row (rows live in 16-lane halves -> width-16 shfl)
            float alpha[8];
            for (int r = 0; r < 8; ++r) {
                float mx = fmaxf(s[0][r], s[1][r]);
                for (int d = 8; d >= 1; d >>= 1) mx = fmaxf(mx, __shfl_xor(mx, d, 16));
                float mnew = fmaxf(mrow[r], mx);
                alpha[r] = exp2f((mrow[r] - mnew) * L2E);
                float p0 = exp2f((s[0][r] - mnew) * L2E);
                float p1 = exp2f((s[1][r] - mnew) * L2E);
                s[0][r] = p0; s[1][r] = p1;
                float ps = p0 + p1;
                for (int d = 8; d >= 1; d >>= 1) ps += __shfl_xor(ps, d, 16);
                lrow[r] = lrow[r] * alpha[r] + ps;
                mrow[r] = mnew;
            }
            // transpose P (C-layout -> A-frag) through per-wave LDS buffer
            for (int nt = 0; nt < 2; ++nt)
                for (int r = 0; r < 8; ++r)
                    pBuf[wave][(r + mofs) * 32 + nt * 16 + (lane & 15)] = (_Float16)s[nt][r];
            v16h pf;
            {
                int m = lane & 15;
                v8h lo = *(const v8h*)(&pBuf[wave][m * 32 + base]);
                v8h hi = *(const v8h*)(&pBuf[wave][m * 32 + 16 + base]);
                for (int i = 0; i < 8; ++i) { pf[i] = lo[i]; pf[8 + i] = hi[i]; }
            }
            // rescale O then accumulate P·V: 8 WMMAs
            for (int j = 0; j < 8; ++j) {
                for (int r = 0; r < 8; ++r) o[j][r] *= alpha[r];
                o[j] = wmma_f16(pf, *(const v16h*)(&vFrag[j][lane * 16]), o[j]);
            }
        }
    }

    // epilogue: normalize and store f16 attn output [b, q, h]
    for (int j = 0; j < 8; ++j)
        for (int r = 0; r < 8; ++r) {
            int q = q0 + r + mofs;
            int head = j * 16 + (lane & 15);
            AO[((size_t)b * SEQ + q) * HD + head] = (_Float16)(o[j][r] / lrow[r]);
        }
}

// ---------------------------------------------------------------------------
// Kernel 3: output projection  out[m,d] = sum_h AO[m,h]*Wo[d,h] + bo[d]
// ---------------------------------------------------------------------------
__global__ __launch_bounds__(256) void oproj_kernel(
    const _Float16* __restrict__ AO, const _Float16* __restrict__ WoF,
    const float* __restrict__ bo, float* __restrict__ out)
{
    const int t    = threadIdx.x;
    const int lane = t & 31;
    const int wave = t >> 5;
    const int m0   = blockIdx.x * 16;
    const int n0   = blockIdx.y * 128 + wave * 16;
    const int base = (lane & 16) ? 8 : 0;

    v8f acc = {0,0,0,0,0,0,0,0};
    int m = m0 + (lane & 15);
    for (int ks = 0; ks < 4; ++ks) {
        v8h lo = *(const v8h*)(AO + (size_t)m * HD + ks * 32 + base);
        v8h hi = *(const v8h*)(AO + (size_t)m * HD + ks * 32 + 16 + base);
        v16h a;
        for (int i = 0; i < 8; ++i) { a[i] = lo[i]; a[8 + i] = hi[i]; }
        v16h bf = *(const v16h*)(WoF + ((size_t)(n0 >> 4) * 4 + ks) * 512 + lane * 16);
        acc = wmma_f16(a, bf, acc);
    }
    int d = n0 + (lane & 15);
    float bias = bo[d];
    int mofs = (lane & 16) ? 8 : 0;
    for (int r = 0; r < 8; ++r)
        out[(size_t)(m0 + r + mofs) * DIN + d] = acc[r] + bias;
}

// ---------------------------------------------------------------------------
extern "C" void kernel_launch(void* const* d_in, const int* in_sizes, int n_in,
                              void* d_out, int out_size, void* d_ws, size_t ws_size,
                              hipStream_t stream) {
    const float* x  = (const float*)d_in[0];
    const float* Wq = (const float*)d_in[1];
    const float* bq = (const float*)d_in[2];
    const float* Wk = (const float*)d_in[3];
    const float* bk = (const float*)d_in[4];
    const float* Wv = (const float*)d_in[5];
    const float* bv = (const float*)d_in[6];
    const float* Wo = (const float*)d_in[7];
    const float* bo = (const float*)d_in[8];
    float* out = (float*)d_out;

    char* ws = (char*)d_ws;
    _Float16* Qh  = (_Float16*)(ws);
    _Float16* Kh  = (_Float16*)(ws + QKV_BYTES);
    _Float16* Vt  = (_Float16*)(ws + 2 * QKV_BYTES);
    _Float16* AO  = (_Float16*)(ws + 3 * QKV_BYTES);
    _Float16* WqF = (_Float16*)(ws + 4 * QKV_BYTES);
    _Float16* WkF = WqF + WFRAG_HALVES;
    _Float16* WvF = WkF + WFRAG_HALVES;
    _Float16* WoF = WvF + WFRAG_HALVES;

    repack_kernel<<<(4 * WFRAG_HALVES) / 256, 256, 0, stream>>>(
        Wq, Wk, Wv, Wo, WqF, WkF, WvF, WoF);
    qkv_kernel<<<MTOT / 16, 256, 0, stream>>>(
        x, bq, bk, bv, WqF, WkF, WvF, Qh, Kh, Vt);
    attn_kernel<<<NB * (SEQ / 128), 256, 0, stream>>>(Qh, Kh, Vt, AO);
    oproj_kernel<<<dim3(MTOT / 16, DIN / 128), 256, 0, stream>>>(AO, WoF, bo, out);
}
